// GATcoarseVAE_9311489098381
// MI455X (gfx1250) — matrix operand order repaired and checked by reading, not compiled, not run on hardware
//
#include <hip/hip_runtime.h>
#include <hip/hip_bf16.h>
#include <math.h>
#include <stdint.h>

// ---------------------------------------------------------------------------
// GAT + VGAE forward for MI455X (gfx1250), wave32 + WMMA bf16 (f32 accumulate)
// All GEMMs consume B pre-transposed ([N,K] row-major) so A- and B-tile
// staging are contiguous 16B/lane copies -> async global->LDS eligible.
// ---------------------------------------------------------------------------

typedef __attribute__((ext_vector_type(16))) __bf16 v16bf;
typedef __attribute__((ext_vector_type(8)))  __bf16 bf16x8;
typedef __attribute__((ext_vector_type(8)))  float  v8f;
typedef __attribute__((__vector_size__(16))) int    v4i32;

#define NN    8192
#define FIN   512
#define H1D   256
#define H2D   8192
#define ALPHA 0.2f
#define NEGI  (-9.0e15f)

#if defined(__HIP_DEVICE_COMPILE__) && __has_builtin(__builtin_amdgcn_global_load_async_to_lds_b128)
#define HAS_ASYNC_LDS 1
#else
#define HAS_ASYNC_LDS 0
#endif

#if defined(__HIP_DEVICE_COMPILE__) && __has_builtin(__builtin_amdgcn_s_wait_asynccnt)
#define WAIT_ASYNC(n) __builtin_amdgcn_s_wait_asynccnt(n)
#else
#define WAIT_ASYNC(n) asm volatile("s_wait_asynccnt %0" ::"i"(n) : "memory")
#endif

#if HAS_ASYNC_LDS
typedef __attribute__((address_space(1))) v4i32* as1_v4i32p;
typedef __attribute__((address_space(3))) v4i32* as3_v4i32p;
static __device__ __forceinline__ void async_cp16(const __bf16* g, __bf16* l) {
  __builtin_amdgcn_global_load_async_to_lds_b128(
      (as1_v4i32p)(uintptr_t)g,
      (as3_v4i32p)(uint32_t)(uintptr_t)l, 0, 0);
}
#endif

static __device__ __forceinline__ v16bf join8(bf16x8 lo, bf16x8 hi) {
  v16bf r;
#pragma unroll
  for (int i = 0; i < 8; ++i) { r[i] = lo[i]; r[i + 8] = hi[i]; }
  return r;
}

// ---------------------------------------------------------------------------
// C[M,N] = A[M,K] * Bsrc^T, with Bsrc given as [N,K] row-major, bf16 in,
// f32 accumulate.  MODE: 1 = elu -> bf16 row-major
//                        2 = f32 row-major
//                        3 = bf16 transposed store (CT is [N,M] row-major)
// Block tile 128x128, K-tile 32, 256 threads = 8 waves (wave tile 32x64).
// ---------------------------------------------------------------------------
template <int MODE>
__global__ __launch_bounds__(256) void wmma_gemm_kernel(
    const __bf16* __restrict__ A, const __bf16* __restrict__ Bsrc,
    float* __restrict__ Cf, __bf16* __restrict__ Cb, int M, int N, int K) {
  constexpr int LDA = 40;  // 32 + 8 pad (bf16 elems)
  __shared__ __bf16 sA[2][128 * LDA];
  __shared__ __bf16 sB[2][128 * LDA];

  const int tid  = threadIdx.x;
  const int lane = tid & 31;
  const int wid  = tid >> 5;
  const int wm   = wid >> 1;   // 0..3
  const int wn   = wid & 1;    // 0..1
  const int h    = lane >> 4;
  const int l16  = lane & 15;
  const int bm0  = blockIdx.y * 128;
  const int bn0  = blockIdx.x * 128;

  const v8f vzero = {0.f, 0.f, 0.f, 0.f, 0.f, 0.f, 0.f, 0.f};
  v8f acc[2][4];
#pragma unroll
  for (int i = 0; i < 2; ++i)
#pragma unroll
    for (int j = 0; j < 4; ++j) acc[i][j] = vzero;

  const int ntiles = K >> 5;

  auto compute = [&](int buf) {
    v16bf bfrag[4];
#pragma unroll
    for (int nt = 0; nt < 4; ++nt) {
      int nn = wn * 64 + nt * 16 + l16;
      const __bf16* p = &sB[buf][nn * LDA + h * 8];
      bfrag[nt] = join8(*(const bf16x8*)p, *(const bf16x8*)(p + 16));
    }
    v16bf afrag[2];
#pragma unroll
    for (int mt = 0; mt < 2; ++mt) {
      int mm = wm * 32 + mt * 16 + l16;
      const __bf16* p = &sA[buf][mm * LDA + h * 8];
      afrag[mt] = join8(*(const bf16x8*)p, *(const bf16x8*)(p + 16));
    }
#pragma unroll
    for (int mt = 0; mt < 2; ++mt)
#pragma unroll
      for (int nt = 0; nt < 4; ++nt)
        acc[mt][nt] = __builtin_amdgcn_wmma_f32_16x16x32_bf16(
            false, afrag[mt], false, bfrag[nt], (short)0, acc[mt][nt], false, false);
  };

#if HAS_ASYNC_LDS
  // --- software-pipelined async global->LDS staging (ASYNCcnt) ---
  auto issue = [&](int t, int buf) {
    int k0 = t << 5;
#pragma unroll
    for (int it = 0; it < 2; ++it) {
      int c   = tid + (it << 8);
      int row = c >> 2;
      int c8  = (c & 3) << 3;
      async_cp16(A    + (size_t)(bm0 + row) * K + k0 + c8, &sA[buf][row * LDA + c8]);
      async_cp16(Bsrc + (size_t)(bn0 + row) * K + k0 + c8, &sB[buf][row * LDA + c8]);
    }
  };
  issue(0, 0);
  for (int t = 0; t < ntiles; ++t) {
    if (t + 1 < ntiles) {
      issue(t + 1, (t + 1) & 1);
      WAIT_ASYNC(4);  // 4 ops of tile t+1 outstanding; tile t landed (in-order)
    } else {
      WAIT_ASYNC(0);
    }
    __syncthreads();
    // L2 prefetch two tiles ahead
    if (t + 2 < ntiles)
      __builtin_prefetch(A + (size_t)(bm0 + (tid & 127)) * K + ((t + 2) << 5), 0, 1);
    compute(t & 1);
    __syncthreads();
  }
#else
  // --- fallback: batched synchronous staging, single buffer ---
  for (int t = 0; t < ntiles; ++t) {
    int k0 = t << 5;
    bf16x8 va[2], vb[2];
#pragma unroll
    for (int it = 0; it < 2; ++it) {
      int c   = tid + (it << 8);
      int row = c >> 2;
      int c8  = (c & 3) << 3;
      va[it] = *(const bf16x8*)(A    + (size_t)(bm0 + row) * K + k0 + c8);
      vb[it] = *(const bf16x8*)(Bsrc + (size_t)(bn0 + row) * K + k0 + c8);
    }
    __syncthreads();  // previous compute done before overwrite
#pragma unroll
    for (int it = 0; it < 2; ++it) {
      int c   = tid + (it << 8);
      int row = c >> 2;
      int c8  = (c & 3) << 3;
      *(bf16x8*)(&sA[0][row * LDA + c8]) = va[it];
      *(bf16x8*)(&sB[0][row * LDA + c8]) = vb[it];
    }
    __syncthreads();
    compute(0);
  }
#endif

  // ---- epilogue: D layout -> row = h*8 + r (consecutive!), col = l16 ----
#pragma unroll
  for (int mt = 0; mt < 2; ++mt)
#pragma unroll
    for (int nt = 0; nt < 4; ++nt) {
      int gm0 = bm0 + wm * 32 + mt * 16 + h * 8;
      int gn  = bn0 + wn * 64 + nt * 16 + l16;
      if (MODE == 3) {
        bf16x8 pk;
#pragma unroll
        for (int r = 0; r < 8; ++r) pk[r] = (__bf16)acc[mt][nt][r];
        *(bf16x8*)(&Cb[(size_t)gn * M + gm0]) = pk;  // CT[n][m], 16B store
      } else {
#pragma unroll
        for (int r = 0; r < 8; ++r) {
          float v   = acc[mt][nt][r];
          size_t ix = (size_t)(gm0 + r) * N + gn;
          if (MODE == 1) {
            float e = v > 0.f ? v : (expf(v) - 1.f);  // elu
            Cb[ix] = (__bf16)e;
          } else {
            Cf[ix] = v;
          }
        }
      }
    }
}

// ---------------------------------------------------------------------------
// packers
// ---------------------------------------------------------------------------
__global__ void f32_to_bf16_kernel(const float* __restrict__ in,
                                   __bf16* __restrict__ out, int n) {
  int i = blockIdx.x * blockDim.x + threadIdx.x;
  if (i < n) out[i] = (__bf16)in[i];
}

// WgatT [H2D, FIN]: WgatT[c*FIN + k] = W_gat[k*H2D + c]
__global__ void pack_wgatT_kernel(const float* __restrict__ W,
                                  __bf16* __restrict__ out) {
  int i = blockIdx.x * blockDim.x + threadIdx.x;
  if (i >= H2D * FIN) return;
  int k = i & (FIN - 1), c = i >> 9;
  out[i] = (__bf16)W[(size_t)k * H2D + c];
}

// WgcT [2*H1, FIN]: row c<256 from W_gc2 col c, else W_gc3 col c-256
__global__ void pack_wgcT_kernel(const float* __restrict__ W2,
                                 const float* __restrict__ W3,
                                 __bf16* __restrict__ out) {
  int i = blockIdx.x * blockDim.x + threadIdx.x;
  if (i >= 2 * H1D * FIN) return;
  int k = i & (FIN - 1), c = i >> 9;
  float v = (c < H1D) ? W2[(size_t)k * H1D + c] : W3[(size_t)k * H1D + (c - H1D)];
  out[i] = (__bf16)v;
}

// f1[i] = sum_j WhT[j*NN+i]*a[j] ; f2[i] = sum_j WhT[j*NN+i]*a[H2D+j]
__global__ __launch_bounds__(256) void f1f2_kernel(
    const __bf16* __restrict__ WhT, const float* __restrict__ a,
    float* __restrict__ f1, float* __restrict__ f2) {
  int i = blockIdx.x * 256 + threadIdx.x;  // node index, coalesced over WhT cols
  float s1 = 0.f, s2 = 0.f;
  for (int j = 0; j < H2D; ++j) {
    float w = (float)WhT[(size_t)j * NN + i];
    s1 += w * a[j];
    s2 += w * a[H2D + j];
  }
  f1[i] = s1;
  f2[i] = s2;
}

// att[i,:] = softmax_j( adj[i,j]>0 ? lrelu(f1[i]+f2[j]) : -9e15 ), bf16 out
__global__ __launch_bounds__(256) void attn_softmax_kernel(
    const float* __restrict__ adj, const float* __restrict__ f1,
    const float* __restrict__ f2, __bf16* __restrict__ att) {
  int row = blockIdx.x, tid = threadIdx.x;
  float fi = f1[row];
  float ev[32];
  float mx = -3.0e38f;
#pragma unroll
  for (int it = 0; it < 32; ++it) {
    int j   = tid + it * 256;
    float e = fi + f2[j];
    e = e >= 0.f ? e : ALPHA * e;
    float v = (adj[(size_t)row * NN + j] > 0.f) ? e : NEGI;
    ev[it] = v;
    mx = fmaxf(mx, v);
  }
  __shared__ float red[256];
  red[tid] = mx;
  __syncthreads();
  for (int s = 128; s > 0; s >>= 1) {
    if (tid < s) red[tid] = fmaxf(red[tid], red[tid + s]);
    __syncthreads();
  }
  float m = red[0];
  __syncthreads();
  float sum = 0.f;
#pragma unroll
  for (int it = 0; it < 32; ++it) {
    ev[it] = expf(ev[it] - m);
    sum += ev[it];
  }
  red[tid] = sum;
  __syncthreads();
  for (int s = 128; s > 0; s >>= 1) {
    if (tid < s) red[tid] += red[tid + s];
    __syncthreads();
  }
  float inv = 1.f / red[0];
#pragma unroll
  for (int it = 0; it < 32; ++it) {
    int j = tid + it * 256;
    att[(size_t)row * NN + j] = (__bf16)(ev[it] * inv);
  }
}

// P[NN, 2*H1] f32 -> mu (f32 out), logvar (f32 out), mu bf16 (for mu@mu^T)
__global__ void split_kernel(const float* __restrict__ P,
                             float* __restrict__ mu_o,
                             float* __restrict__ lv_o,
                             __bf16* __restrict__ mu_b) {
  int i = blockIdx.x * blockDim.x + threadIdx.x;
  if (i >= NN * 2 * H1D) return;
  int r = i / (2 * H1D), c = i % (2 * H1D);
  float v = P[i];
  if (c < H1D) {
    mu_o[r * H1D + c] = v;
    mu_b[r * H1D + c] = (__bf16)v;
  } else {
    lv_o[r * H1D + (c - H1D)] = v;
  }
}

// ---------------------------------------------------------------------------
// launcher
// ---------------------------------------------------------------------------
extern "C" void kernel_launch(void* const* d_in, const int* in_sizes, int n_in,
                              void* d_out, int out_size, void* d_ws, size_t ws_size,
                              hipStream_t stream) {
  (void)in_sizes; (void)n_in; (void)out_size; (void)ws_size;
  const float* x     = (const float*)d_in[0];
  const float* adj   = (const float*)d_in[1];
  // d_in[2] = adj0 (unused)
  const float* W_gat = (const float*)d_in[3];
  const float* a_gat = (const float*)d_in[4];
  const float* W_gc2 = (const float*)d_in[5];
  const float* W_gc3 = (const float*)d_in[6];

  float* out     = (float*)d_out;
  float* adj_rec = out;                        // [NN, NN]
  float* mu_out  = out + (size_t)NN * NN;      // [NN, H1]
  float* lv_out  = mu_out + (size_t)NN * H1D;  // [NN, H1]

  char* ws = (char*)d_ws;
  const size_t MB = 1024ull * 1024ull;
  __bf16* WhT_bf  = (__bf16*)(ws + 0);                   // 128 MB [H2D, NN] = Wh^T
  __bf16* att_bf  = (__bf16*)(ws + 128 * MB);            // 128 MB [NN, NN]
  __bf16* h1_bf   = (__bf16*)(ws + 256 * MB);            // 128 MB [NN, H2D]
  __bf16* x_bf    = (__bf16*)(ws + 384 * MB);            //   8 MB [NN, FIN]
  __bf16* WgatT   = (__bf16*)(ws + 392 * MB);            //   8 MB [H2D, FIN]
  __bf16* WgcT    = (__bf16*)(ws + 400 * MB);            // 0.5 MB [2*H1, FIN]
  __bf16* xwT_bf  = (__bf16*)(ws + 401 * MB);            //   8 MB [2*H1, NN] = xW^T
  float*  f1      = (float*)(ws + 409 * MB);             //  32 KB
  float*  f2      = (float*)(ws + 409 * MB + 65536);     //  32 KB
  // overlay onto dead att region after GEMM2:
  float*  P_f32   = (float*)(ws + 128 * MB);             //  16 MB [NN, 2*H1]
  __bf16* mu_bf   = (__bf16*)(ws + 128 * MB + 16 * MB);  //   4 MB [NN, H1]

  const int T = 256;

  // 1) pack operands to bf16 (weights transposed)
  f32_to_bf16_kernel<<<(NN * FIN + T - 1) / T, T, 0, stream>>>(x, x_bf, NN * FIN);
  pack_wgatT_kernel<<<(H2D * FIN + T - 1) / T, T, 0, stream>>>(W_gat, WgatT);
  pack_wgcT_kernel<<<(2 * H1D * FIN + T - 1) / T, T, 0, stream>>>(W_gc2, W_gc3, WgcT);

  // 2) Wh^T = (x @ W_gat)^T            [8192,512] x [512,8192] -> bf16 [H2D,NN]
  wmma_gemm_kernel<3><<<dim3(H2D / 128, NN / 128), T, 0, stream>>>(
      x_bf, WgatT, nullptr, WhT_bf, NN, H2D, FIN);

  // 3) f1/f2 row scores (coalesced reads of Wh^T columns)
  f1f2_kernel<<<NN / 256, T, 0, stream>>>(WhT_bf, a_gat, f1, f2);

  // 4) masked leaky-relu softmax -> att (bf16)
  attn_softmax_kernel<<<NN, T, 0, stream>>>(adj, f1, f2, att_bf);

  // 5) hidden1 = elu(att @ Wh)         [8192,8192] x [8192,8192] -> bf16, B = Wh^T
  wmma_gemm_kernel<1><<<dim3(H2D / 128, NN / 128), T, 0, stream>>>(
      att_bf, WhT_bf, nullptr, h1_bf, NN, H2D, NN);

  // 6) xW^T = (x @ [W_gc2|W_gc3])^T    [8192,512] x [512,512] -> bf16 [512,NN]
  wmma_gemm_kernel<3><<<dim3((2 * H1D) / 128, NN / 128), T, 0, stream>>>(
      x_bf, WgcT, nullptr, xwT_bf, NN, 2 * H1D, FIN);

  // 7) P = hidden1 @ xW                [8192,8192] x [8192,512] -> f32, B = xW^T
  wmma_gemm_kernel<2><<<dim3((2 * H1D) / 128, NN / 128), T, 0, stream>>>(
      h1_bf, xwT_bf, P_f32, nullptr, NN, 2 * H1D, H2D);

  // 8) split into mu / logvar outputs (+ mu in bf16)
  split_kernel<<<(NN * 2 * H1D + T - 1) / T, T, 0, stream>>>(P_f32, mu_out, lv_out, mu_bf);

  // 9) adj_rec = mu @ mu^T             [8192,256] x [256,8192] -> f32, B = mu
  wmma_gemm_kernel<2><<<dim3(NN / 128, NN / 128), T, 0, stream>>>(
      mu_bf, mu_bf, adj_rec, nullptr, NN, NN, H1D);
}
